// AFNB_33947421508409
// MI455X (gfx1250) — matrix operand order repaired and checked by reading, not compiled, not run on hardware
//
#include <hip/hip_runtime.h>
#include <hip/hip_bf16.h>

// ---------------------------------------------------------------------------
// AFNB (asymmetric non-local) for MI455X / gfx1250.
// All GEMMs run through v_wmma_f32_16x16x32_bf16 (wave32, 16x16x32 tiles).
// value path uses psp(conv(x)) == conv(psp(x)) to shrink that GEMM to 110 cols.
// ---------------------------------------------------------------------------

typedef __attribute__((ext_vector_type(16))) __bf16        v16bf;
typedef __attribute__((ext_vector_type(8)))  float         v8f;
typedef __attribute__((ext_vector_type(8)))  unsigned int  v8u;

#define TILE_M 256
#define TILE_N 128
#define TILE_K 32
#define LDT    40      // padded LDS row stride in ushorts (80B: keeps B128 stores aligned)

__device__ __forceinline__ unsigned short f2bf(float f) {
  unsigned int u = __builtin_bit_cast(unsigned int, f);
  unsigned int r = u + 0x7FFFu + ((u >> 16) & 1u);   // round-to-nearest-even
  return (unsigned short)(r >> 16);
}
__device__ __forceinline__ float bf2f(unsigned short h) {
  unsigned int u = ((unsigned int)h) << 16;
  return __builtin_bit_cast(float, u);
}
__device__ __forceinline__ unsigned int pk2bf(float lo, float hi) {
  return (unsigned int)f2bf(lo) | ((unsigned int)f2bf(hi) << 16);
}
__device__ __forceinline__ float ldval(const float* p, long i) { return p[i]; }
__device__ __forceinline__ float ldval(const unsigned short* p, long i) { return bf2f(p[i]); }

// ---------------------------------------------------------------------------
// Generic strided GEMM:  C(m,n) = epilogue( sum_k A(m,k)*B(k,n) )
//   A(m,k) = A[b*Abs + m*Ars + k*Acs]   (fp32 or bf16, converted into LDS)
//   B(k,n) = (k<ksplit ? B0[b*Bbs0 + k*Brs + n*Bcs]
//                      : B1[b*Bbs1 + (k-ksplit)*Brs + n*Bcs])
//   MODE 0: fp32 store of acc*alpha
//   MODE 1: bf16 store of acc + bias[m]        (bias may be null)
//   MODE 2: bf16 store of relu(bn(acc+bias))   (inference BN)
//   MODE 3: fp32 store of bn(acc+bias)
// Block tile 256x128, 8 wave32 waves in 4(M) x 2(N); wave tile 64x64 ->
// 16 accumulators, 16 v_wmma per K-step of 32.
// ---------------------------------------------------------------------------
struct GemmP {
  const void* A; const void* B0; const void* B1; void* C;
  const float* bias; const float* g; const float* be; const float* mean; const float* var;
  long Abs, Ars, Acs;
  long Bbs0, Bbs1, Brs, Bcs;
  long Cbs, Crs, Ccs;
  int  M, N, K, ksplit;
  float alpha;
};

template <int AF32, int BF32, int MODE>
__global__ __launch_bounds__(256) void gemm_wmma_bf16(GemmP p) {
  __shared__ unsigned short As[TILE_M * LDT];   // [m][k]
  __shared__ unsigned short Bs[TILE_N * LDT];   // [n][k]  (B tile stored transposed)

  const int tid  = threadIdx.x;
  const int lane = tid & 31;
  const int wave = tid >> 5;
  const int wm   = wave >> 1;          // 0..3  -> 64 rows each
  const int wn   = wave & 1;           // 0..1  -> 64 cols each
  const int half = lane >> 4;
  const int l15  = lane & 15;

  const int m0 = blockIdx.y * TILE_M;
  const int n0 = blockIdx.x * TILE_N;
  const int b  = blockIdx.z;

  v8f acc[4][4];
  const v8f vzero = {0.f,0.f,0.f,0.f,0.f,0.f,0.f,0.f};
#pragma unroll
  for (int mi = 0; mi < 4; ++mi)
#pragma unroll
    for (int ni = 0; ni < 4; ++ni) acc[mi][ni] = vzero;

  const bool aCont = (p.Acs == 1);
  const int KT = (p.K + TILE_K - 1) / TILE_K;
  for (int kt = 0; kt < KT; ++kt) {
    const int k0 = kt * TILE_K;
    const bool fullK = (k0 + TILE_K) <= p.K;
    const bool fullA = fullK && ((m0 + TILE_M) <= p.M);
    const bool fullB = fullK && ((n0 + TILE_N) <= p.N);

    // ---------------- stage A tile (256 x 32) ----------------
    if (aCont && fullA) {
      // vector fast path: 8 contiguous k per chunk, 4 chunks/thread
#pragma unroll
      for (int c = 0; c < 4; ++c) {
        int chunk = tid + c * 256;
        int i = chunk >> 2, j = (chunk & 3) * 8;
        long off = (long)b * p.Abs + (long)(m0 + i) * p.Ars + (k0 + j);
        uint4 u;
        if (AF32) {
          const float4* fp = (const float4*)((const float*)p.A + off);
          float4 f0 = fp[0], f1 = fp[1];
          u.x = pk2bf(f0.x, f0.y); u.y = pk2bf(f0.z, f0.w);
          u.z = pk2bf(f1.x, f1.y); u.w = pk2bf(f1.z, f1.w);
          if (fullK && (k0 + TILE_K + TILE_K) <= p.K)
            __builtin_prefetch((const float*)p.A + off + TILE_K, 0, 1);
        } else {
          u = *(const uint4*)((const unsigned short*)p.A + off);
          if (fullK && (k0 + TILE_K + TILE_K) <= p.K)
            __builtin_prefetch((const unsigned short*)p.A + off + TILE_K, 0, 1);
        }
        *(uint4*)&As[i * LDT + j] = u;
      }
    } else {
      // guarded scalar path (strided / edge tiles)
      for (int e = 0; e < (TILE_M * TILE_K) / 256; ++e) {
        int idx = tid + e * 256;
        int i = idx >> 5, j = idx & 31;
        int gm = m0 + i, gk = k0 + j;
        unsigned short v = 0;
        if (gm < p.M && gk < p.K) {
          long off = (long)b * p.Abs + (long)gm * p.Ars + (long)gk * p.Acs;
          v = AF32 ? f2bf(((const float*)p.A)[off])
                   : ((const unsigned short*)p.A)[off];
        }
        As[i * LDT + j] = v;
      }
    }

    // ---------------- stage B tile (32 x 128), transposed into LDS ----------------
    if (fullB && p.Brs == 1) {
      // k-contiguous source (e.g. context^T): straight B128 copy rows
#pragma unroll
      for (int c = 0; c < 2; ++c) {
        int chunk = tid + c * 256;
        int nidx = chunk >> 2, j = (chunk & 3) * 8;
        int gk = k0 + j;
        const void* Bp; long base;
        if (gk < p.ksplit) { Bp = p.B0; base = (long)b * p.Bbs0 + gk; }
        else               { Bp = p.B1; base = (long)b * p.Bbs1 + (gk - p.ksplit); }
        long off = base + (long)(n0 + nidx) * p.Bcs;
        uint4 u;
        if (BF32) {
          const float4* fp = (const float4*)((const float*)Bp + off);
          float4 f0 = fp[0], f1 = fp[1];
          u.x = pk2bf(f0.x, f0.y); u.y = pk2bf(f0.z, f0.w);
          u.z = pk2bf(f1.x, f1.y); u.w = pk2bf(f1.z, f1.w);
        } else {
          u = *(const uint4*)((const unsigned short*)Bp + off);
        }
        *(uint4*)&Bs[nidx * LDT + j] = u;
      }
    } else if (fullB && p.Bcs == 1) {
      // n-contiguous source (activations): vector load, b16 scatter into LDS
#pragma unroll
      for (int c = 0; c < 2; ++c) {
        int chunk = tid + c * 256;
        int j = chunk >> 4, nb = (chunk & 15) * 8;
        int gk = k0 + j;
        const void* Bp; long base;
        if (gk < p.ksplit) { Bp = p.B0; base = (long)b * p.Bbs0 + (long)gk * p.Brs; }
        else               { Bp = p.B1; base = (long)b * p.Bbs1 + (long)(gk - p.ksplit) * p.Brs; }
        long off = base + (n0 + nb);
        unsigned short vals[8];
        if (BF32) {
          const float4* fp = (const float4*)((const float*)Bp + off);
          float4 f0 = fp[0], f1 = fp[1];
          vals[0] = f2bf(f0.x); vals[1] = f2bf(f0.y); vals[2] = f2bf(f0.z); vals[3] = f2bf(f0.w);
          vals[4] = f2bf(f1.x); vals[5] = f2bf(f1.y); vals[6] = f2bf(f1.z); vals[7] = f2bf(f1.w);
          if (fullK && (k0 + TILE_K + TILE_K) <= p.K)
            __builtin_prefetch((const float*)Bp + off + (long)TILE_K * p.Brs, 0, 1);
        } else {
          uint4 u = *(const uint4*)((const unsigned short*)Bp + off);
          vals[0] = (unsigned short)u.x; vals[1] = (unsigned short)(u.x >> 16);
          vals[2] = (unsigned short)u.y; vals[3] = (unsigned short)(u.y >> 16);
          vals[4] = (unsigned short)u.z; vals[5] = (unsigned short)(u.z >> 16);
          vals[6] = (unsigned short)u.w; vals[7] = (unsigned short)(u.w >> 16);
          if (fullK && (k0 + TILE_K + TILE_K) <= p.K)
            __builtin_prefetch((const unsigned short*)Bp + off + (long)TILE_K * p.Brs, 0, 1);
        }
#pragma unroll
        for (int e = 0; e < 8; ++e) Bs[(nb + e) * LDT + j] = vals[e];
      }
    } else {
      for (int e = 0; e < (TILE_N * TILE_K) / 256; ++e) {
        int idx = tid + e * 256;
        int n = idx >> 5, j = idx & 31;
        int gn = n0 + n, gk = k0 + j;
        unsigned short v = 0;
        if (gn < p.N && gk < p.K) {
          long off;
          const void* Bp;
          if (gk < p.ksplit) { Bp = p.B0; off = (long)b * p.Bbs0 + (long)gk * p.Brs + (long)gn * p.Bcs; }
          else               { Bp = p.B1; off = (long)b * p.Bbs1 + (long)(gk - p.ksplit) * p.Brs + (long)gn * p.Bcs; }
          v = BF32 ? f2bf(((const float*)Bp)[off])
                   : ((const unsigned short*)Bp)[off];
        }
        Bs[n * LDT + j] = v;
      }
    }
    __syncthreads();

    // ---- fragments (documented 16-bit 16x32 A layout) + 16 WMMA ----
    v16bf afr[4], bfr[4];
#pragma unroll
    for (int mi = 0; mi < 4; ++mi) {
      const unsigned short* base = &As[(wm * 64 + mi * 16 + l15) * LDT];
      v8u u;
#pragma unroll
      for (int v = 0; v < 8; ++v) {
        int kk = ((v >> 2) << 4) + (half << 3) + ((v & 3) << 1);
        u[v] = *(const unsigned int*)(base + kk);
      }
      afr[mi] = __builtin_bit_cast(v16bf, u);
    }
#pragma unroll
    for (int ni = 0; ni < 4; ++ni) {
      const unsigned short* base = &Bs[(wn * 64 + ni * 16 + l15) * LDT];
      v8u u;
#pragma unroll
      for (int v = 0; v < 8; ++v) u[v] = *(const unsigned int*)(base + (((v >> 2) << 4) + (half << 3) + ((v & 3) << 1)));
      bfr[ni] = __builtin_bit_cast(v16bf, u);
    }
#pragma unroll
    for (int mi = 0; mi < 4; ++mi)
#pragma unroll
      for (int ni = 0; ni < 4; ++ni)
        acc[mi][ni] = __builtin_amdgcn_wmma_f32_16x16x32_bf16(
            false, afr[mi], false, bfr[ni], (short)0, acc[mi][ni], false, false);
    __syncthreads();
  }

  // ---- epilogue + store (C 16x16 layout: VGPR r -> M = half*8 + r) ----
#pragma unroll
  for (int mi = 0; mi < 4; ++mi) {
#pragma unroll
    for (int r = 0; r < 8; ++r) {
      int gm = m0 + wm * 64 + mi * 16 + half * 8 + r;
      if (gm >= p.M) continue;
      float badd = 0.f, scale = 1.f, shift = 0.f;
      if (MODE == 1) { badd = p.bias ? p.bias[gm] : 0.f; }
      if (MODE == 2 || MODE == 3) {
        scale = p.g[gm] * rsqrtf(p.var[gm] + 1e-5f);
        shift = p.be[gm] - (p.mean[gm] - p.bias[gm]) * scale;  // bn(acc+bias)
      }
      long crow = (long)b * p.Cbs + (long)gm * p.Crs;
#pragma unroll
      for (int ni = 0; ni < 4; ++ni) {
        int gn = n0 + wn * 64 + ni * 16 + l15;
        if (gn < p.N) {
          float x = acc[mi][ni][r] * p.alpha;
          long coff = crow + (long)gn * p.Ccs;
          if (MODE == 0) {
            ((float*)p.C)[coff] = x;
          } else if (MODE == 1) {
            ((unsigned short*)p.C)[coff] = f2bf(x + badd);
          } else if (MODE == 2) {
            ((unsigned short*)p.C)[coff] = f2bf(fmaxf(x * scale + shift, 0.f));
          } else {
            ((float*)p.C)[coff] = x * scale + shift;
          }
        }
      }
    }
  }
}

// ---------------------------------------------------------------------------
// fp32 -> bf16 bulk convert
// ---------------------------------------------------------------------------
__global__ void cvt_f32_bf16(const float* __restrict__ in,
                             unsigned short* __restrict__ out, long n) {
  long i = (long)blockIdx.x * blockDim.x + threadIdx.x;
  if (i < n) out[i] = f2bf(in[i]);
}

// ---------------------------------------------------------------------------
// Spatial pyramid pooling: in (B*C, H, W) -> out bf16 (B*C, 112)
// cells: s=1 (1), s=3 (9), s=6 (36), s=8 (64) -> 110 valid + 2 zero pad.
// One wave32 per pooled cell, shuffle reduction.
// ---------------------------------------------------------------------------
template <typename T>
__global__ void psp_pool(const T* __restrict__ in, unsigned short* __restrict__ out,
                         int total, int HH, int WW) {
  int job  = blockIdx.x * 8 + (threadIdx.x >> 5);
  int lane = threadIdx.x & 31;
  if (job >= total) return;
  int loc = job % 110;
  int bc  = job / 110;
  int s, t;
  if (loc == 0)       { s = 1; t = 0; }
  else if (loc < 10)  { s = 3; t = loc - 1; }
  else if (loc < 46)  { s = 6; t = loc - 10; }
  else                { s = 8; t = loc - 46; }
  int bi = t / s, bj = t % s;
  int bh = HH / s, bw = WW / s;
  const T* src = in + (long)bc * HH * WW;
  float sum = 0.f;
  for (int e = lane; e < bh * bw; e += 32) {
    int hh = bi * bh + e / bw;
    int ww = bj * bw + e % bw;
    sum += ldval(src, (long)hh * WW + ww);
  }
#pragma unroll
  for (int o = 16; o > 0; o >>= 1) sum += __shfl_xor(sum, o, 32);
  if (lane == 0) {
    long base = (long)bc * 112;
    out[base + loc] = f2bf(sum / (float)(bh * bw));
    if (loc == 0) { out[base + 110] = 0; out[base + 111] = 0; }
  }
}

// ---------------------------------------------------------------------------
// Row softmax over 110 logits (fp32 in) -> bf16 out (112 cols, pad zeroed)
// One wave32 per row.
// ---------------------------------------------------------------------------
__global__ void softmax110(const float* __restrict__ sim,
                           unsigned short* __restrict__ out, int rows) {
  int row  = blockIdx.x * 8 + (threadIdx.x >> 5);
  int lane = threadIdx.x & 31;
  if (row >= rows) return;
  const float* rp = sim + (long)row * 112;
  float v[4];
  float mx = -3.4e38f;
#pragma unroll
  for (int it = 0; it < 4; ++it) {
    int j = lane + 32 * it;
    v[it] = (j < 110) ? rp[j] : -3.4e38f;
    mx = fmaxf(mx, v[it]);
  }
#pragma unroll
  for (int o = 16; o > 0; o >>= 1) mx = fmaxf(mx, __shfl_xor(mx, o, 32));
  float sum = 0.f;
#pragma unroll
  for (int it = 0; it < 4; ++it) {
    int j = lane + 32 * it;
    v[it] = (j < 110) ? __expf(v[it] - mx) : 0.f;
    sum += v[it];
  }
#pragma unroll
  for (int o = 16; o > 0; o >>= 1) sum += __shfl_xor(sum, o, 32);
  float inv = 1.f / sum;
  unsigned short* op = out + (long)row * 112;
#pragma unroll
  for (int it = 0; it < 4; ++it) {
    int j = lane + 32 * it;
    if (j < 112) op[j] = (j < 110) ? f2bf(v[it] * inv) : (unsigned short)0;
  }
}

// ---------------------------------------------------------------------------
// Host-side orchestration
// ---------------------------------------------------------------------------
extern "C" void kernel_launch(void* const* d_in, const int* in_sizes, int n_in,
                              void* d_out, int out_size, void* d_ws, size_t ws_size,
                              hipStream_t stream) {
  (void)in_sizes; (void)n_in; (void)out_size; (void)ws_size;
  const int Bc = 4, CL = 1024, CH = 2048, CK = 256, CV = 256, CO = 2048, H = 72, W = 72;
  const int N = H * W;          // 5184
  const int P = 112;            // 110 pooled cells, padded to 112

  const float* low  = (const float*)d_in[0];
  const float* high = (const float*)d_in[1];
  const float* wk = (const float*)d_in[2],  *bk  = (const float*)d_in[3];
  const float* gk = (const float*)d_in[4],  *bek = (const float*)d_in[5];
  const float* mk = (const float*)d_in[6],  *vk  = (const float*)d_in[7];
  const float* wq = (const float*)d_in[8],  *bq  = (const float*)d_in[9];
  const float* gq = (const float*)d_in[10], *beq = (const float*)d_in[11];
  const float* mq = (const float*)d_in[12], *vq  = (const float*)d_in[13];
  const float* wv = (const float*)d_in[14], *bv  = (const float*)d_in[15];
  const float* ww = (const float*)d_in[16], *bw  = (const float*)d_in[17];
  const float* wf = (const float*)d_in[18], *bf  = (const float*)d_in[19];
  const float* gf = (const float*)d_in[20], *bef = (const float*)d_in[21];
  const float* mf = (const float*)d_in[22], *vf  = (const float*)d_in[23];
  float* out = (float*)d_out;

  // workspace carve-up
  char* wp = (char*)d_ws;
  auto alloc = [&](size_t bytes) { void* r = wp; wp += (bytes + 255) & ~(size_t)255; return r; };
  unsigned short* high_bf = (unsigned short*)alloc((size_t)Bc * CH * N * 2); // 85 MB
  unsigned short* lowp    = (unsigned short*)alloc((size_t)Bc * CL * P * 2);
  unsigned short* val_bf  = (unsigned short*)alloc((size_t)Bc * P * CV * 2); // [b][loc][cv]
  unsigned short* key_act = (unsigned short*)alloc((size_t)Bc * CK * N * 2);
  unsigned short* keyp    = (unsigned short*)alloc((size_t)Bc * CK * P * 2);
  unsigned short* q_bf    = (unsigned short*)alloc((size_t)Bc * CK * N * 2);
  float*          sim_f   = (float*)        alloc((size_t)Bc * N * P * 4);
  unsigned short* sim_bf  = (unsigned short*)alloc((size_t)Bc * N * P * 2);
  unsigned short* ctx_bf  = (unsigned short*)alloc((size_t)Bc * N * CV * 2); // [b][p][cv]
  unsigned short* ctxw_bf = (unsigned short*)alloc((size_t)Bc * CO * N * 2); // 85 MB

  auto tm = [](int x) { return (x + TILE_M - 1) / TILE_M; };
  auto tn = [](int x) { return (x + TILE_N - 1) / TILE_N; };

  // 1) high -> bf16 once (reused by query conv + fusion conv; stays L2-resident)
  {
    long n = (long)Bc * CH * N;
    cvt_f32_bf16<<<(unsigned)((n + 255) / 256), 256, 0, stream>>>(high, high_bf, n);
  }
  // 2) psp(low) (value path: pool before conv — linear ops commute)
  {
    int total = Bc * CL * 110;
    psp_pool<float><<<(total + 7) / 8, 256, 0, stream>>>(low, lowp, total, H, W);
  }
  // 3) value = wv * psp(low) + bv   -> bf16 [b][loc][cv]
  {
    GemmP p{};
    p.A = wv; p.B0 = p.B1 = lowp; p.C = val_bf; p.bias = bv;
    p.Abs = 0; p.Ars = CL; p.Acs = 1;
    p.Bbs0 = p.Bbs1 = (long)CL * P; p.Brs = P; p.Bcs = 1;
    p.Cbs = (long)P * CV; p.Crs = 1; p.Ccs = CV;
    p.M = CV; p.N = P; p.K = CL; p.ksplit = CL; p.alpha = 1.f;
    gemm_wmma_bf16<1,0,1><<<dim3(tn(P), tm(CV), Bc), 256, 0, stream>>>(p);
  }
  // 4) key_act = relu(bn(wk*low + bk))  -> bf16 (B,CK,N)
  {
    GemmP p{};
    p.A = wk; p.B0 = p.B1 = low; p.C = key_act;
    p.bias = bk; p.g = gk; p.be = bek; p.mean = mk; p.var = vk;
    p.Abs = 0; p.Ars = CL; p.Acs = 1;
    p.Bbs0 = p.Bbs1 = (long)CL * N; p.Brs = N; p.Bcs = 1;
    p.Cbs = (long)CK * N; p.Crs = N; p.Ccs = 1;
    p.M = CK; p.N = N; p.K = CL; p.ksplit = CL; p.alpha = 1.f;
    gemm_wmma_bf16<1,1,2><<<dim3(tn(N), tm(CK), Bc), 256, 0, stream>>>(p);
  }
  // 5) keyp = psp(key_act)
  {
    int total = Bc * CK * 110;
    psp_pool<unsigned short><<<(total + 7) / 8, 256, 0, stream>>>(key_act, keyp, total, H, W);
  }
  // 6) query = relu(bn(wq*high + bq)) -> bf16 (B,CK,N)
  {
    GemmP p{};
    p.A = wq; p.B0 = p.B1 = high_bf; p.C = q_bf;
    p.bias = bq; p.g = gq; p.be = beq; p.mean = mq; p.var = vq;
    p.Abs = 0; p.Ars = CH; p.Acs = 1;
    p.Bbs0 = p.Bbs1 = (long)CH * N; p.Brs = N; p.Bcs = 1;
    p.Cbs = (long)CK * N; p.Crs = N; p.Ccs = 1;
    p.M = CK; p.N = N; p.K = CH; p.ksplit = CH; p.alpha = 1.f;
    gemm_wmma_bf16<1,0,2><<<dim3(tn(N), tm(CK), Bc), 256, 0, stream>>>(p);
  }
  // 7) sim = (Q^T K) * CK^-0.5  -> fp32 (B,N,112), only 110 cols written
  {
    GemmP p{};
    p.A = q_bf; p.B0 = p.B1 = keyp; p.C = sim_f;
    p.Abs = (long)CK * N; p.Ars = 1; p.Acs = N;     // A = Q^T (pixel-major)
    p.Bbs0 = p.Bbs1 = (long)CK * P; p.Brs = P; p.Bcs = 1;
    p.Cbs = (long)N * P; p.Crs = P; p.Ccs = 1;
    p.M = N; p.N = 110; p.K = CK; p.ksplit = CK; p.alpha = 0.0625f; // 256^-0.5
    gemm_wmma_bf16<0,0,0><<<dim3(tn(110), tm(N), Bc), 256, 0, stream>>>(p);
  }
  // 8) softmax over 110, write bf16 with zeroed pad cols
  {
    int rows = Bc * N;
    softmax110<<<(rows + 7) / 8, 256, 0, stream>>>(sim_f, sim_bf, rows);
  }
  // 9) context = softmax(sim) * value -> bf16 [b][p][cv]  (pad K handled by zeros)
  {
    GemmP p{};
    p.A = sim_bf; p.B0 = p.B1 = val_bf; p.C = ctx_bf; p.bias = nullptr;
    p.Abs = (long)N * P; p.Ars = P; p.Acs = 1;
    p.Bbs0 = p.Bbs1 = (long)P * CV; p.Brs = CV; p.Bcs = 1;
    p.Cbs = (long)N * CV; p.Crs = CV; p.Ccs = 1;
    p.M = N; p.N = CV; p.K = P; p.ksplit = P; p.alpha = 1.f;
    gemm_wmma_bf16<0,0,1><<<dim3(tn(CV), tm(N), Bc), 256, 0, stream>>>(p);
  }
  // 10) ctxw = ww * context^T + bw -> bf16 (B,CO,N)
  {
    GemmP p{};
    p.A = ww; p.B0 = p.B1 = ctx_bf; p.C = ctxw_bf; p.bias = bw;
    p.Abs = 0; p.Ars = CV; p.Acs = 1;
    p.Bbs0 = p.Bbs1 = (long)N * CV; p.Brs = 1; p.Bcs = CV;  // B = context^T (k-contig)
    p.Cbs = (long)CO * N; p.Crs = N; p.Ccs = 1;
    p.M = CO; p.N = N; p.K = CV; p.ksplit = CV; p.alpha = 1.f;
    gemm_wmma_bf16<1,0,1><<<dim3(tn(N), tm(CO), Bc), 256, 0, stream>>>(p);
  }
  // 11) out = bn(wf * concat(ctxw, high) + bf)  — split-K B source, fp32 out
  {
    GemmP p{};
    p.A = wf; p.B0 = ctxw_bf; p.B1 = high_bf; p.C = out;
    p.bias = bf; p.g = gf; p.be = bef; p.mean = mf; p.var = vf;
    p.Abs = 0; p.Ars = CO + CH; p.Acs = 1;
    p.Bbs0 = (long)CO * N; p.Bbs1 = (long)CH * N; p.Brs = N; p.Bcs = 1;
    p.Cbs = (long)CO * N; p.Crs = N; p.Ccs = 1;
    p.M = CO; p.N = N; p.K = CO + CH; p.ksplit = CO; p.alpha = 1.f;
    gemm_wmma_bf16<1,0,3><<<dim3(tn(N), tm(CO), Bc), 256, 0, stream>>>(p);
  }
}
